// DecoderCrossAttn_2594160247255
// MI455X (gfx1250) — compile-verified, matrix operand order checked by reading, not run on hardware
//
#include <hip/hip_runtime.h>
#include <hip/hip_bf16.h>

typedef __attribute__((ext_vector_type(16))) _Float16 v16h;
typedef __attribute__((ext_vector_type(8)))  _Float16 v8h;
typedef __attribute__((ext_vector_type(2)))  _Float16 v2h;
typedef __attribute__((ext_vector_type(8)))  float    v8f;

#define DEV __device__ __forceinline__

static constexpr int Bb   = 8;
static constexpr int SEQ  = 1024;   // N == M
static constexpr int Dm   = 768;
static constexpr int NH   = 12;
static constexpr int DH   = 64;
static constexpr int TOK  = Bb * SEQ;   // 8192

DEV v16h mk16(v8h lo, v8h hi) {
  v16h r;
#pragma unroll
  for (int i = 0; i < 8; ++i) { r[i] = lo[i]; r[i + 8] = hi[i]; }
  return r;
}

// Load one 16x32 f16 WMMA fragment (A-layout, or B-layout from a transposed
// [n][k] tile). rowbase points at this lane's row (16B aligned); grp = lane>>4.
DEV v16h ldfrag(const _Float16* rowbase, int grp) {
  v8h lo = *(const v8h*)(rowbase + grp * 8);
  v8h hi = *(const v8h*)(rowbase + 16 + grp * 8);
  return mk16(lo, hi);
}

DEV v8f wmma32(v16h a, v16h b, v8f c) {
  return __builtin_amdgcn_wmma_f32_16x16x32_f16(false, a, false, b, (short)0, c,
                                                false, false);
}

// CDNA5 async copy global->LDS, 16B per lane, tracked by ASYNCcnt.
// Generic addresses of __shared__ objects carry the LDS byte offset in
// addr[31:0] (ISA 10.2 aperture rule), which is exactly what VDST wants.
DEV void async_b128(const void* g, void* l) {
  unsigned loff = (unsigned)(size_t)l;
  unsigned long long ga = (unsigned long long)(size_t)g;
  asm volatile("global_load_async_to_lds_b128 %0, %1, off"
               :: "v"(loff), "v"(ga) : "memory");
}
DEV void wait_async0() { asm volatile("s_wait_asynccnt 0" ::: "memory"); }

// 16-lane max reduction step via DPP (stays inside each 16-lane row).
template <int CTRL>
DEV float dppmax(float v) {
  int t = __builtin_amdgcn_mov_dpp(__builtin_bit_cast(int, v), CTRL, 0xf, 0xf,
                                   true);
  return fmaxf(v, __builtin_bit_cast(float, t));
}
DEV float rowmax16(float v) {
  v = dppmax<0xB1>(v);    // quad_perm(1,0,3,2)  : xor 1
  v = dppmax<0x4E>(v);    // quad_perm(2,3,0,1)  : xor 2
  v = dppmax<0x141>(v);   // row_half_mirror     : pairs across quads in each 8
  v = dppmax<0x140>(v);   // row_mirror          : pairs across the two 8s
  return v;
}

// ---------------------------------------------------------------- convert
__global__ void cvt_f32_f16(const float* __restrict__ in,
                            _Float16* __restrict__ out, int n) {
  int i = blockIdx.x * blockDim.x + threadIdx.x;
  if (i < n) out[i] = (_Float16)in[i];
}

// ---------------------------------------------------------------- GEMM
// C[Mtot x 768] = A[Mtot x 768] (f16) @ W[768 x 768] (f16) + bias(f32)
// Block tile 128x128, K-step 64. 8 waves, each 32(M) x 64(N).
template <bool OUT16>
__global__ void __launch_bounds__(256)
gemm_kernel(const _Float16* __restrict__ A, const _Float16* __restrict__ W,
            const float* __restrict__ bias, void* __restrict__ Cout) {
  __shared__ __align__(16) _Float16 As[128 * 64];   // [m][k]
  __shared__ __align__(16) _Float16 Bs[128 * 64];   // transposed: [n][k]

  const int tid = threadIdx.x, lane = tid & 31, w = tid >> 5;
  const int ln = lane & 15, grp = lane >> 4;
  const int m0 = blockIdx.x * 128, n0 = blockIdx.y * 128;
  const int wm = (w & 3) * 32;      // wave M offset within block
  const int wn = (w >> 2) * 64;     // wave N offset within block

  v8f acc[2][4] = {};

  for (int k0 = 0; k0 < Dm; k0 += 64) {
    // --- stage A tile 128x64 via async 16B copies (straight copy)
#pragma unroll
    for (int i = 0; i < 4; ++i) {
      int c = tid + i * 256;            // 1024 chunks of v8h
      int row = c >> 3, seg = c & 7;
      async_b128(&A[(size_t)(m0 + row) * Dm + k0 + seg * 8],
                 &As[row * 64 + seg * 8]);
    }
    // --- stage W tile transposed: read [k][n] coalesced, write [n][k]
#pragma unroll
    for (int i = 0; i < 16; ++i) {
      int p = tid + i * 256;            // 4096 u32 pairs
      int kk = p >> 6, n = (p & 63) * 2;
      v2h v = *(const v2h*)&W[(size_t)(k0 + kk) * Dm + n0 + n];
      Bs[(n + 0) * 64 + kk] = v[0];
      Bs[(n + 1) * 64 + kk] = v[1];
    }
    if (k0 + 64 < Dm)
      __builtin_prefetch(&A[(size_t)(m0 + (tid >> 1)) * Dm + k0 + 64], 0, 1);
    wait_async0();
    __syncthreads();

#pragma unroll
    for (int kc = 0; kc < 2; ++kc) {
      v16h af[2], bf[4];
#pragma unroll
      for (int i = 0; i < 2; ++i)
        af[i] = ldfrag(&As[(wm + i * 16 + ln) * 64 + kc * 32], grp);
#pragma unroll
      for (int j = 0; j < 4; ++j)
        bf[j] = ldfrag(&Bs[(wn + j * 16 + ln) * 64 + kc * 32], grp);
#pragma unroll
      for (int i = 0; i < 2; ++i)
#pragma unroll
        for (int j = 0; j < 4; ++j) acc[i][j] = wmma32(af[i], bf[j], acc[i][j]);
    }
    __syncthreads();
  }

  // epilogue: C-layout -> lane holds col n, VGPR r holds row 8*grp+r
#pragma unroll
  for (int i = 0; i < 2; ++i)
#pragma unroll
    for (int j = 0; j < 4; ++j) {
      int ngl = n0 + wn + j * 16 + ln;
      float bv = bias[ngl];
#pragma unroll
      for (int r = 0; r < 8; ++r) {
        int mg = m0 + wm + i * 16 + grp * 8 + r;
        float vout = acc[i][j][r] + bv;
        if (OUT16)
          ((_Float16*)Cout)[(size_t)mg * Dm + ngl] = (_Float16)vout;
        else
          ((float*)Cout)[(size_t)mg * Dm + ngl] = vout;
      }
    }
}

// ---------------------------------------------------------------- RoPE 2D
// t: [TOK][768] f16 (heads inline). One thread per rotation pair.
__global__ void rope_kernel(_Float16* __restrict__ t,
                            const int* __restrict__ pos) {
  int tid = blockIdx.x * blockDim.x + threadIdx.x;  // TOK*NH*2*16 threads
  int jj = tid & 15;
  int half = (tid >> 4) & 1;
  int h = (tid >> 5) % NH;
  int tok = tid / (32 * NH);
  if (tok >= TOK) return;

  int p = pos[tok * 2 + half];                 // half0 -> row, half1 -> col
  float ang = (float)p * __expf(-0.28782313662425572f * (float)jj); // ln(100)/16
  float s, c;
  __sincosf(ang, &s, &c);

  size_t base = (size_t)tok * Dm + h * DH + half * 32 + jj;
  float t0 = (float)t[base];
  float t1 = (float)t[base + 16];
  t[base]      = (_Float16)(t0 * c - t1 * s);
  t[base + 16] = (_Float16)(t1 * c + t0 * s);
}

// ---------------------------------------------------------------- attention
// grid (SEQ/64, B*NH); 128 threads = 4 waves x 16 query rows.
// Q is pre-scaled by dh^-0.5 * log2(e), so softmax runs in exp2 domain.
// Row sums are accumulated with an extra WMMA against a ones matrix
// (P x 1 -> every column holds the row sum, already in C-layout indexing).
__global__ void __launch_bounds__(128)
attn_kernel(const _Float16* __restrict__ Q, const _Float16* __restrict__ K,
            const _Float16* __restrict__ V, _Float16* __restrict__ O) {
  __shared__ __align__(16) _Float16 Kt[32 * 64];   // [key][d]
  __shared__ __align__(16) _Float16 Vt[64 * 32];   // transposed [d][key]
  __shared__ __align__(16) _Float16 Pl[4 * 16 * 32];

  const int b = blockIdx.y / NH, h = blockIdx.y % NH;
  const int q0 = blockIdx.x * 64;
  const int tid = threadIdx.x, lane = tid & 31, w = tid >> 5;
  const int ln = lane & 15, grp = lane >> 4;

  // Q fragments (16x64 A-matrix = 2 chunks of 16x32), loaded once;
  // fold dh^-0.5 * log2(e) = 0.125 * 1.4426950 into Q (f16, tiny rel. error).
  const _Float16 qscale = (_Float16)0.18033688f;
  const _Float16* qp = Q + ((size_t)(b * SEQ + q0 + w * 16 + ln)) * Dm + h * DH;
  v16h qf[2];
#pragma unroll
  for (int c = 0; c < 2; ++c) {
    qf[c] = ldfrag(qp + c * 32, grp);
#pragma unroll
    for (int i = 0; i < 16; ++i) qf[c][i] = qf[c][i] * qscale;
  }

  v16h onesf;
#pragma unroll
  for (int i = 0; i < 16; ++i) onesf[i] = (_Float16)1.0f;

  v8f o[4] = {};
  v8f osum = {};
  float rmax[8];
#pragma unroll
  for (int r = 0; r < 8; ++r) rmax[r] = -1e30f;

  for (int j0 = 0; j0 < SEQ; j0 += 32) {
    // stage K tile [32 keys][64 d] via async 16B copies
#pragma unroll
    for (int i = 0; i < 2; ++i) {
      int p = tid + i * 128;                // 256 v8h chunks
      int key = p >> 3, seg = p & 7;
      async_b128(&K[((size_t)(b * SEQ + j0 + key)) * Dm + h * DH + seg * 8],
                 &Kt[key * 64 + seg * 8]);
    }
    // stage V tile transposed -> [d][key] (manual, it permutes)
#pragma unroll
    for (int i = 0; i < 8; ++i) {
      int p = tid + i * 128;                // 1024 u32 pairs
      int key = p >> 5, d = (p & 31) * 2;
      v2h v = *(const v2h*)&V[((size_t)(b * SEQ + j0 + key)) * Dm + h * DH + d];
      Vt[(d + 0) * 32 + key] = v[0];
      Vt[(d + 1) * 32 + key] = v[1];
    }
    if (j0 + 32 < SEQ)
      __builtin_prefetch(&K[((size_t)(b * SEQ + j0 + 32 + (tid >> 2))) * Dm +
                            h * DH], 0, 1);
    wait_async0();
    __syncthreads();

    // scores (log2 domain): two 16x16 tiles, K-dim = dh = 64 (2 wmma each)
    v8f s[2];
#pragma unroll
    for (int t = 0; t < 2; ++t) {
      const _Float16* krow = Kt + (t * 16 + ln) * 64;
      v8f z = {};
      z = wmma32(qf[0], ldfrag(krow, grp), z);
      z = wmma32(qf[1], ldfrag(krow + 32, grp), z);
      s[t] = z;
    }

    // online softmax; row r lives in lanes sharing (lane>>4): DPP max
    float alpha[8];
#pragma unroll
    for (int r = 0; r < 8; ++r) {
      float lm = rowmax16(fmaxf(s[0][r], s[1][r]));
      float nm = fmaxf(rmax[r], lm);
      alpha[r] = __builtin_amdgcn_exp2f(rmax[r] - nm);
      rmax[r] = nm;
      s[0][r] = __builtin_amdgcn_exp2f(s[0][r] - nm);
      s[1][r] = __builtin_amdgcn_exp2f(s[1][r] - nm);
    }
#pragma unroll
    for (int nt = 0; nt < 4; ++nt)
#pragma unroll
      for (int r = 0; r < 8; ++r) o[nt][r] *= alpha[r];
#pragma unroll
    for (int r = 0; r < 8; ++r) osum[r] *= alpha[r];

    // C-layout -> A-layout via per-wave LDS patch
    _Float16* pw = Pl + w * 512;
#pragma unroll
    for (int t = 0; t < 2; ++t)
#pragma unroll
      for (int r = 0; r < 8; ++r)
        pw[(grp * 8 + r) * 32 + t * 16 + ln] = (_Float16)s[t][r];
    asm volatile("s_wait_dscnt 0" ::: "memory");

    v16h pf = ldfrag(pw + ln * 32, grp);
#pragma unroll
    for (int nt = 0; nt < 4; ++nt) {
      v16h vf = ldfrag(Vt + (nt * 16 + ln) * 32, grp);
      o[nt] = wmma32(pf, vf, o[nt]);
    }
    osum = wmma32(pf, onesf, osum);     // row sums, C-layout, f32 accum
    __syncthreads();
  }

  // normalize and store f16
#pragma unroll
  for (int nt = 0; nt < 4; ++nt)
#pragma unroll
    for (int r = 0; r < 8; ++r) {
      int row = q0 + w * 16 + grp * 8 + r;
      int col = h * DH + nt * 16 + ln;
      O[((size_t)(b * SEQ + row)) * Dm + col] = (_Float16)(o[nt][r] / osum[r]);
    }
}

// ---------------------------------------------------------------- launch
extern "C" void kernel_launch(void* const* d_in, const int* in_sizes, int n_in,
                              void* d_out, int out_size, void* d_ws,
                              size_t ws_size, hipStream_t stream) {
  const float* x  = (const float*)d_in[0];
  const float* y  = (const float*)d_in[1];
  const int* pq   = (const int*)d_in[2];
  const int* pkv  = (const int*)d_in[3];
  const float* Wq = (const float*)d_in[4];
  const float* bq = (const float*)d_in[5];
  const float* Wk = (const float*)d_in[6];
  const float* bk = (const float*)d_in[7];
  const float* Wv = (const float*)d_in[8];
  const float* bv = (const float*)d_in[9];
  const float* Wo = (const float*)d_in[10];
  const float* bo = (const float*)d_in[11];

  const size_t ACT = (size_t)TOK * Dm;   // 6,291,456 halves
  const size_t WSZ = (size_t)Dm * Dm;    //   589,824 halves

  char* ws = (char*)d_ws;
  _Float16* xh  = (_Float16*)ws;                ws += ACT * 2;
  _Float16* yh  = (_Float16*)ws;                ws += ACT * 2;
  _Float16* Wqh = (_Float16*)ws;                ws += WSZ * 2;
  _Float16* Wkh = (_Float16*)ws;                ws += WSZ * 2;
  _Float16* Wvh = (_Float16*)ws;                ws += WSZ * 2;
  _Float16* Woh = (_Float16*)ws;                ws += WSZ * 2;
  _Float16* Qh  = (_Float16*)ws;                ws += ACT * 2;
  _Float16* Kh  = (_Float16*)ws;                ws += ACT * 2;
  _Float16* Vh  = (_Float16*)ws;                ws += ACT * 2;
  _Float16* Oh  = (_Float16*)ws;                ws += ACT * 2;

  // 1) f32 -> f16
  cvt_f32_f16<<<(int)((ACT + 255) / 256), 256, 0, stream>>>(x, xh, (int)ACT);
  cvt_f32_f16<<<(int)((ACT + 255) / 256), 256, 0, stream>>>(y, yh, (int)ACT);
  cvt_f32_f16<<<(int)((WSZ + 255) / 256), 256, 0, stream>>>(Wq, Wqh, (int)WSZ);
  cvt_f32_f16<<<(int)((WSZ + 255) / 256), 256, 0, stream>>>(Wk, Wkh, (int)WSZ);
  cvt_f32_f16<<<(int)((WSZ + 255) / 256), 256, 0, stream>>>(Wv, Wvh, (int)WSZ);
  cvt_f32_f16<<<(int)((WSZ + 255) / 256), 256, 0, stream>>>(Wo, Woh, (int)WSZ);

  // 2) Q/K/V projections (WMMA, bias fused, f16 out)
  dim3 ggrid(TOK / 128, Dm / 128);
  gemm_kernel<true><<<ggrid, 256, 0, stream>>>(xh, Wqh, bq, Qh);
  gemm_kernel<true><<<ggrid, 256, 0, stream>>>(yh, Wkh, bk, Kh);
  gemm_kernel<true><<<ggrid, 256, 0, stream>>>(yh, Wvh, bv, Vh);

  // 3) 2D RoPE on Q and K
  int rthreads = TOK * NH * 32;
  rope_kernel<<<rthreads / 256, 256, 0, stream>>>(Qh, pq);
  rope_kernel<<<rthreads / 256, 256, 0, stream>>>(Kh, pkv);

  // 4) flash attention (WMMA)
  attn_kernel<<<dim3(SEQ / 64, Bb * NH), 128, 0, stream>>>(Qh, Kh, Vh, Oh);

  // 5) output projection, f32 out
  gemm_kernel<false><<<ggrid, 256, 0, stream>>>(Oh, Woh, bo, (float*)d_out);
}
